// conv3dDeformable_33002528703000
// MI455X (gfx1250) — matrix operand order repaired and checked by reading, not compiled, not run on hardware
//
#include <hip/hip_runtime.h>
#include <hip/hip_bf16.h>

// ---- CDNA5 WMMA vector types --------------------------------------------
typedef __bf16 v16bf __attribute__((ext_vector_type(16)));
typedef float  v8f   __attribute__((ext_vector_type(8)));

// round-to-nearest-even float -> bf16 bits
static __device__ __forceinline__ unsigned short f2bf(float f) {
    union { float f; unsigned int u; } v; v.f = f;
    unsigned int u = v.u;
    return (unsigned short)((u + 0x7FFFu + ((u >> 16) & 1u)) >> 16);
}

// ---------------------------------------------------------------------------
// Weight prep: repack w[oc][ic][kd][kh][kw] (oc-stride 32*27) into per-tap
// WMMA A fragments (bf16), laid out so each lane reads 32 contiguous bytes:
//   frag[((tile*27 + tap)*32 + lane)*16 + e]
// A 16x32 bf16, contiguous k order used here (same as B): m = lane&15, g = lane>>4,
//   K(e) = 16*g + e
// ---------------------------------------------------------------------------
__global__ void wprep_kernel(const float* __restrict__ w,
                             unsigned short* __restrict__ frag) {
    const int blk  = blockIdx.x;       // tile*27 + tap, grid = 4*27
    const int tap  = blk % 27;
    const int tile = blk / 27;
    const int lane = threadIdx.x;      // 0..31
    const int m = lane & 15, g = lane >> 4;
    const int oc = tile * 16 + m;
    unsigned short tmp[16];
#pragma unroll
    for (int e = 0; e < 16; ++e) {
        int ic = 16 * g + e;
        tmp[e] = f2bf(w[(oc * 32 + ic) * 27 + tap]);
    }
    unsigned short* dst = frag + ((size_t)(tile * 27 + tap) * 32 + lane) * 16;
#pragma unroll
    for (int e = 0; e < 16; ++e) dst[e] = tmp[e];
}

// ---------------------------------------------------------------------------
// Implicit-GEMM 3x3x3 conv (stride 1, pad 1), Cin=32, Cout=64, vol 64^3, B=2.
// Block (256 thr = 8 waves) owns one full (b,d,h) W-row (64 positions), all
// 64 output channels.
// LDS: bf16 halo patch [row = kdh*66 + wIdx][ic=0..31]  (9*66*32 = 38 KB).
// Wave(w): ocTile = w>>1, posPair = w&1 (posTiles 2p, 2p+1).
// Per tap: ONE A fragment load feeds TWO WMMAs (two independent
// accumulators) -> 54 v_wmma per wave, 48 B/lane operand traffic per WMMA.
// ---------------------------------------------------------------------------
template <bool IN_BF16>
__global__ void __launch_bounds__(256)
conv_tap_wmma_kernel(const void* __restrict__ in,               // f32 or bf16 [B][32][64^3]
                     const unsigned short* __restrict__ wfrag,  // [4][27][32][16] bf16
                     const float* __restrict__ bias,            // [64] or nullptr
                     float* __restrict__ out) {                 // f32 [B][64][64^3]
    __shared__ __align__(32) unsigned short patch[594 * 32];    // 38,016 B

    const int bi  = blockIdx.x;        // grid = 2*64*64 = 8192
    const int h   = bi & 63;
    const int d   = (bi >> 6) & 63;
    const int b   = bi >> 12;
    const int tid = threadIdx.x;

    const float*          inF = (const float*)in;
    const unsigned short* inH = (const unsigned short*)in;

    // cooperative halo fill: f = ic*594 + row  (consecutive tid -> consecutive
    // wIdx -> coalesced global loads along W)
    for (int f = tid; f < 594 * 32; f += 256) {
        int ic   = f / 594;
        int row  = f - ic * 594;          // row = kdh*66 + wIdx
        int kdh  = row / 66;
        int wIdx = row - kdh * 66;
        int kd   = kdh / 3;
        int kh   = kdh - kd * 3;
        int zin  = d + kd - 1;
        int yin  = h + kh - 1;
        int xin  = wIdx - 1;
        unsigned short hv = 0;
        if ((unsigned)zin < 64u && (unsigned)yin < 64u && (unsigned)xin < 64u) {
            size_t idx = (((size_t)(b * 32 + ic) * 64 + zin) * 64 + yin) * 64 + xin;
            hv = IN_BF16 ? inH[idx] : f2bf(inF[idx]);
        }
        patch[row * 32 + ic] = hv;
    }
    __syncthreads();

    const int wave    = tid >> 5;
    const int lane    = tid & 31;
    const int ocTile  = wave >> 1;
    const int posPair = wave & 1;      // posTiles 2*posPair, 2*posPair+1
    const int g       = lane >> 4;     // K half-block for B; M half for C/D
    const int n       = lane & 15;     // N (position) for B/C/D
    const int p0      = posPair * 32;  // W offset of first pos tile

    v8f acc0 = {};
    v8f acc1 = {};
    // A fragment pointer in v16bf units: ((ocTile*27 + tap)*32 + lane)
    const v16bf* afrag = reinterpret_cast<const v16bf*>(wfrag)
                       + (size_t)ocTile * 27 * 32 + lane;

#pragma unroll
    for (int kd = 0; kd < 3; ++kd) {
#pragma unroll
        for (int kh = 0; kh < 3; ++kh) {
            const int kdh = kd * 3 + kh;
#pragma unroll
            for (int kw = 0; kw < 3; ++kw) {
                const int tap = kdh * 3 + kw;
                v16bf a = afrag[(size_t)tap * 32];
                // B 32x16: lane n = column, g selects K 0..15 / 16..31,
                // 16 contiguous bf16 in one LDS row.
                const int row0 = kdh * 66 + p0 + n + kw;
                v16bf b0 = *reinterpret_cast<const v16bf*>(&patch[row0 * 32 + g * 16]);
                v16bf b1 = *reinterpret_cast<const v16bf*>(&patch[(row0 + 16) * 32 + g * 16]);
                acc0 = __builtin_amdgcn_wmma_f32_16x16x32_bf16(
                           false, a, false, b0, (short)0, acc0, false, false);
                acc1 = __builtin_amdgcn_wmma_f32_16x16x32_bf16(
                           false, a, false, b1, (short)0, acc1, false, false);
            }
        }
    }

    // C/D layout: lanes 0-15 -> M = r, lanes 16-31 -> M = 8 + r; N = lane&15
#pragma unroll
    for (int j = 0; j < 2; ++j) {
        const int wOut = p0 + j * 16 + n;
#pragma unroll
        for (int r = 0; r < 8; ++r) {
            int oc = ocTile * 16 + (g ? (8 + r) : r);
            float v = j ? acc1[r] : acc0[r];
            if (bias) v += bias[oc];
            out[(((size_t)(b * 64 + oc) * 64 + d) * 64 + h) * 64 + wOut] = v;
        }
    }
}

// ---------------------------------------------------------------------------
// Trilinear grid sample, zeros padding, align_corners=True, with the
// reference's meshgrid-'xy' broadcast quirk and off_d := off_h bug:
//   ix = (off_h + lin(d) + 1)*0.5*63   (W axis)
//   iy = (off_w + lin(h) + 1)*0.5*63   (H axis)
//   iz = (off_h + lin(w) + 1)*0.5*63   (D axis)
// offs layout: f32 [B][64][64^3] (ch 0..31 = off_w, 32..63 = off_h)
// ---------------------------------------------------------------------------
__global__ void __launch_bounds__(256)
grid_sample_kernel(const float* __restrict__ x,        // [2][32][64^3]
                   const float* __restrict__ offs,     // [2][64][64^3]
                   unsigned short* __restrict__ samp)  // bf16 [2][32][64^3]
{
    const int f  = blockIdx.x * 256 + threadIdx.x;     // 16,777,216 total
    const int wq = f & 63;
    const int hq = (f >> 6) & 63;
    const int dq = (f >> 12) & 63;
    const int c  = (f >> 18) & 31;
    const int b  = f >> 23;

    const size_t sp      = ((size_t)dq * 64 + hq) * 64 + wq;
    const size_t offBase = (size_t)b * 64 * 262144 + sp;
    const float offW = offs[offBase + (size_t)c * 262144];
    const float offH = offs[offBase + (size_t)(32 + c) * 262144];

    const float inv63 = 1.0f / 63.0f;
    const float lind = -1.f + 2.f * (float)dq * inv63;
    const float linh = -1.f + 2.f * (float)hq * inv63;
    const float linw = -1.f + 2.f * (float)wq * inv63;

    const float ix = ((offH + lind) + 1.f) * 0.5f * 63.f;
    const float iy = ((offW + linh) + 1.f) * 0.5f * 63.f;
    const float iz = ((offH + linw) + 1.f) * 0.5f * 63.f;

    const float x0f = floorf(ix), y0f = floorf(iy), z0f = floorf(iz);
    const float tx = ix - x0f, ty = iy - y0f, tz = iz - z0f;
    const int x0 = (int)x0f, y0 = (int)y0f, z0 = (int)z0f;

    const float* vol = x + (size_t)(b * 32 + c) * 262144;
    float outv = 0.f;
#pragma unroll
    for (int dz = 0; dz < 2; ++dz) {
#pragma unroll
        for (int dy = 0; dy < 2; ++dy) {
#pragma unroll
            for (int dx = 0; dx < 2; ++dx) {
                const int xc = x0 + dx, yc = y0 + dy, zc = z0 + dz;
                const float wgt = (dx ? tx : 1.f - tx)
                                * (dy ? ty : 1.f - ty)
                                * (dz ? tz : 1.f - tz);
                const bool inb = ((unsigned)xc < 64u) && ((unsigned)yc < 64u)
                              && ((unsigned)zc < 64u);
                const int xcc = xc < 0 ? 0 : (xc > 63 ? 63 : xc);
                const int ycc = yc < 0 ? 0 : (yc > 63 ? 63 : yc);
                const int zcc = zc < 0 ? 0 : (zc > 63 ? 63 : zc);
                float val = vol[((size_t)zcc * 64 + ycc) * 64 + xcc];
                outv += wgt * (inb ? val : 0.f);
            }
        }
    }
    samp[(size_t)(b * 32 + c) * 262144 + sp] = f2bf(outv);
}

// ---------------------------------------------------------------------------
extern "C" void kernel_launch(void* const* d_in, const int* in_sizes, int n_in,
                              void* d_out, int out_size, void* d_ws, size_t ws_size,
                              hipStream_t stream) {
    const float* x     = (const float*)d_in[0];  // (2,32,64,64,64)
    const float* w_off = (const float*)d_in[1];  // (96,32,3,3,3) — only oc 0..63 used
    const float* w_reg = (const float*)d_in[2];  // (64,32,3,3,3)
    const float* b_reg = (const float*)d_in[3];  // (64,)
    float* out = (float*)d_out;                  // (2,64,64,64,64) f32

    // workspace layout (bytes)
    char* ws = (char*)d_ws;
    float*          offs    = (float*)ws;                                  // 134,217,728
    unsigned short* fragOff = (unsigned short*)(ws + 134217728);           //     110,592 (pad 131072)
    unsigned short* fragReg = (unsigned short*)(ws + 134217728 + 131072);  //     110,592 (pad 131072)
    unsigned short* samp    = (unsigned short*)(ws + 134217728 + 262144);  //  33,554,432

    // 1) weight repack to WMMA A fragments (both tensors have oc-stride 32*27)
    wprep_kernel<<<108, 32, 0, stream>>>(w_off, fragOff);
    wprep_kernel<<<108, 32, 0, stream>>>(w_reg, fragReg);

    // 2) offset conv (only the 64 live channels), bf16 WMMA, f32 out
    conv_tap_wmma_kernel<false><<<8192, 256, 0, stream>>>(x, fragOff, nullptr, offs);

    // 3) deformable trilinear sampling -> bf16 volume
    grid_sample_kernel<<<65536, 256, 0, stream>>>(x, offs, samp);

    // 4) regular conv + bias, bf16 WMMA, f32 out
    conv_tap_wmma_kernel<true><<<8192, 256, 0, stream>>>(samp, fragReg, b_reg, out);
}
